// DPPConv2d_42537356099776
// MI455X (gfx1250) — matrix operand order repaired and checked by reading, not compiled, not run on hardware
//
#include <hip/hip_runtime.h>
#include <hip/hip_bf16.h>

typedef __attribute__((ext_vector_type(16))) __bf16 v16bf;
typedef __attribute__((ext_vector_type(8)))  float  v8f;
typedef __attribute__((ext_vector_type(4)))  unsigned u32x4;
typedef __attribute__((ext_vector_type(8)))  unsigned u32x8;

#define BS   32
#define CIN  128
#define COUT 128
#define HW   4096   // 64*64
#define HIDDEN 33
#define PPAT 4
#define KTOT 1152   // CIN * 9

// CDNA5 async global->LDS copy (16B per active lane), GVS addressing:
//   mem = SADDR64 + zext(VOFF32), LDS dest = wave LDS base + LDSOFF32
__device__ __forceinline__ void async_copy_b128(unsigned lds_off, unsigned voff,
                                                const void* base) {
  asm volatile("global_load_async_to_lds_b128 %0, %1, %2"
               :: "v"(lds_off), "v"(voff), "s"(base)
               : "memory");
}
__device__ __forceinline__ void wait_async0() {
  asm volatile("s_wait_asynccnt 0" ::: "memory");
}

// CDNA5 Tensor Data Mover: 2D tile (tile_w elems x tile_h rows, 2B elems)
// from global (row stride = row_stride elems) into LDS at lds_addr, with
// LDS padding of 4 DWORDs after every 16 DWORDs (64B row -> 80B pitch).
__device__ __forceinline__ void tdm_load_2d_bf16(unsigned lds_addr,
                                                 const void* gptr,
                                                 unsigned tile_w, unsigned tile_h,
                                                 unsigned row_stride) {
  const unsigned long long ga = (unsigned long long)(uintptr_t)gptr;
  u32x4 g0;
  g0[0] = 1u;                                  // count=1, user descriptor
  g0[1] = lds_addr;                            // D#.lds_addr (bytes)
  g0[2] = (unsigned)ga;                        // global_addr[31:0]
  g0[3] = (unsigned)(ga >> 32) | (2u << 30);   // global_addr[56:32] | type=2
  u32x8 g1;
  g1[0] = (1u << 16)                           // data_size = 2 bytes
        | (1u << 20)                           // pad_enable
        | (3u << 22)                           // pad_interval: 16 DWORDs
        | (3u << 25);                          // pad_amount: 4 DWORDs (16B)
  g1[1] = 0u;                                  // tensor_dim0 = 0x40000000 (lo16 in [31:16] = 0)
  g1[2] = 0x4000u;                             // tensor_dim0 hi16; tensor_dim1 lo16 = 0
  g1[3] = 0x4000u | (tile_w << 16);            // tensor_dim1 hi16; tile_dim0
  g1[4] = tile_h;                              // tile_dim1; tile_dim2 = 0 (2D)
  g1[5] = row_stride;                          // tensor_dim0_stride lo32 (elems)
  g1[6] = 0u;                                  // stride hi16; dim1_stride lo16
  g1[7] = 0u;
  asm volatile("tensor_load_to_lds %0, %1" :: "s"(g0), "s"(g1) : "memory");
}

// ---------------- Kernel 1: global avg pool (NCHW f32) ---------------------
__global__ __launch_bounds__(256) void pool_kernel(
    const float* __restrict__ x, float* __restrict__ pooled) {
  const int bc = blockIdx.x;
  const float* src = x + (size_t)bc * HW;
  float s = 0.f;
  for (int i = threadIdx.x; i < HW; i += 256) s += src[i];
  __shared__ float red[256];
  red[threadIdx.x] = s;
  __syncthreads();
  for (int off = 128; off > 0; off >>= 1) {
    if (threadIdx.x < off) red[threadIdx.x] += red[threadIdx.x + off];
    __syncthreads();
  }
  if (threadIdx.x == 0) pooled[bc] = red[0] * (1.0f / HW);
}

// ---------------- Kernel 2: NCHW f32 -> NHWC bf16 transpose ----------------
__global__ __launch_bounds__(256) void transpose_convert_kernel(
    const float* __restrict__ x, __bf16* __restrict__ xb) {
  const int b  = blockIdx.x >> 6;
  const int p0 = (blockIdx.x & 63) << 6;
  const int t  = threadIdx.x;
  __shared__ __bf16 tile[64][130];
  for (int it = 0; it < 32; ++it) {
    const int idx = it * 256 + t;
    const int c  = idx >> 6;
    const int px = idx & 63;
    tile[px][c] = (__bf16)x[((size_t)(b * CIN + c)) * HW + p0 + px];
  }
  __syncthreads();
  for (int it = 0; it < 32; ++it) {
    const int idx = it * 256 + t;
    const int px = idx >> 7;
    const int c  = idx & 127;
    xb[((size_t)(b * HW + p0 + px)) * CIN + c] = tile[px][c];
  }
}

// ---------------- Kernel 3: MLP + softmax attention ------------------------
__global__ __launch_bounds__(128) void attention_kernel(
    const float* __restrict__ pooled, const float* __restrict__ w1,
    const float* __restrict__ w2, const float* __restrict__ b2,
    float* __restrict__ attn) {
  const int b = blockIdx.x;
  const int t = threadIdx.x;
  __shared__ float pl[CIN];
  __shared__ float hh[HIDDEN];
  pl[t] = pooled[b * CIN + t];
  __syncthreads();
  if (t < HIDDEN) {
    float s = 0.f;
    for (int i = 0; i < CIN; ++i) s += pl[i] * w1[t * CIN + i];
    hh[t] = s > 0.f ? s : 0.f;
  }
  __syncthreads();
  float l[PPAT];
  #pragma unroll
  for (int p = 0; p < PPAT; ++p) {
    const int r = p * COUT + t;
    float s = b2[r];
    for (int j = 0; j < HIDDEN; ++j) s += hh[j] * w2[r * HIDDEN + j];
    l[p] = s * 2.0f;                           // /TEMP, TEMP=0.5
  }
  float mx = l[0];
  #pragma unroll
  for (int p = 1; p < PPAT; ++p) mx = fmaxf(mx, l[p]);
  float e[PPAT], sum = 0.f;
  #pragma unroll
  for (int p = 0; p < PPAT; ++p) { e[p] = __expf(l[p] - mx); sum += e[p]; }
  const float inv = 1.0f / sum;
  #pragma unroll
  for (int p = 0; p < PPAT; ++p) attn[(b * PPAT + p) * COUT + t] = e[p] * inv;
}

// ---------------- Kernel 4: binary-step mask over (p, kh, kw) --------------
__global__ __launch_bounds__(256) void mask_kernel(
    const float* __restrict__ w, const float* __restrict__ thr,
    float* __restrict__ mask) {
  const int q = blockIdx.x;
  const int p = q / 9, kpos = q % 9;
  float s = 0.f;
  for (int oi = threadIdx.x; oi < COUT * CIN; oi += 256)
    s += fabsf(w[(size_t)(p * COUT * CIN + oi) * 9 + kpos]);
  __shared__ float red[256];
  red[threadIdx.x] = s;
  __syncthreads();
  for (int off = 128; off > 0; off >>= 1) {
    if (threadIdx.x < off) red[threadIdx.x] += red[threadIdx.x + off];
    __syncthreads();
  }
  if (threadIdx.x == 0) {
    float ts = 0.f;
    for (int o = 0; o < COUT; ++o) ts += thr[p * COUT + o];
    const float z = red[0] * (1.0f / (COUT * CIN)) - ts * (1.0f / COUT);
    mask[q] = (z > 0.f) ? 1.0f : 0.0f;         // step(sigmoid(z)-0.5) == (z>0)
  }
}

// ---------------- Kernel 5: per-sample aggregated bf16 weights -------------
// layout agg[b][o][kpos][i]  (GEMM-K = kpos*128 + i, contiguous 32-chunks)
__global__ __launch_bounds__(256) void aggregate_kernel(
    const float* __restrict__ w, const float* __restrict__ attn,
    const float* __restrict__ mask, __bf16* __restrict__ agg) {
  const int b = blockIdx.x >> 7;
  const int o = blockIdx.x & 127;
  float at[PPAT];
  #pragma unroll
  for (int p = 0; p < PPAT; ++p) at[p] = attn[(b * PPAT + p) * COUT + o];
  __shared__ float mk[PPAT * 9];
  if (threadIdx.x < PPAT * 9) mk[threadIdx.x] = mask[threadIdx.x];
  __syncthreads();
  __bf16* dst = agg + (size_t)(b * COUT + o) * KTOT;
  for (int idx = threadIdx.x; idx < KTOT; idx += 256) {
    const int kpos = idx >> 7;
    const int i = idx & 127;
    float s = 0.f;
    #pragma unroll
    for (int p = 0; p < PPAT; ++p)
      s += w[(size_t)(((p * COUT + o) * CIN + i)) * 9 + kpos] * mk[p * 9 + kpos] * at[p];
    dst[idx] = (__bf16)s;
  }
}

// ---------------- Kernel 6: implicit-GEMM conv via WMMA bf16 ---------------
// block: 256 thr = 8 waves; one sample, all 128 Cout, 8x8 pixel tile.
// A chunk staged by TDM (tensor_load_to_lds), halo by async-to-LDS.
__global__ __launch_bounds__(256) void conv_wmma_kernel(
    const __bf16* __restrict__ xb, const __bf16* __restrict__ agg,
    float* __restrict__ out) {
  const int bid  = blockIdx.x;
  const int b    = bid >> 6;
  const int tile = bid & 63;
  const int th0  = (tile >> 3) << 3;
  const int tw0  = (tile & 7) << 3;

  const int tid  = threadIdx.x;
  const int wid  = tid >> 5;
  const int lane = tid & 31;
  const int hi   = lane >> 4;
  const int mlo  = lane & 15;

  // 80-byte pitches: 16B aligned, conflict-free (bank stride 20); As pitch
  // reproduced by TDM pad (4 DWORDs after every 16 DWORDs).
  __shared__ __bf16 As[COUT][40];              // 128 o x 32 k used (10.0 KB)
  __shared__ __bf16 halo[10][10][40];          // 100 px x 32 ch used (8.0 KB)

  v8f acc[4];
  #pragma unroll
  for (int i = 0; i < 4; ++i) acc[i] = (v8f){};

  const __bf16* xbB  = xb  + (size_t)b * HW * CIN;
  const __bf16* aggB = agg + (size_t)b * COUT * KTOT;
  const unsigned asBase = (unsigned)(uintptr_t)&As[0][0];

  for (int ic0 = 0; ic0 < CIN; ic0 += 32) {
    // ---- stage halo [10x10 px][32 ch] via async-to-LDS (32B per slot) ----
    if (tid < 200) {
      const int pix  = tid >> 1;
      const int half = tid & 1;
      const int py = pix / 10, px = pix % 10;
      const int gy = th0 - 1 + py;
      const int gx = tw0 - 1 + px;
      const unsigned lds = (unsigned)(uintptr_t)&halo[py][px][half * 16];
      if ((gy >= 0) & (gy < 64) & (gx >= 0) & (gx < 64)) {
        const unsigned voff =
            (unsigned)((((gy << 6) + gx) * CIN + ic0 + half * 16) * 2);
        async_copy_b128(lds, voff, xbB);
        async_copy_b128(lds + 16, voff + 16, xbB);
      } else {
        uint4 z = {0u, 0u, 0u, 0u};
        uint4* d = reinterpret_cast<uint4*>(&halo[py][px][half * 16]);
        d[0] = z;
        d[1] = z;
      }
    }
    wait_async0();                             // each wave's halo slots in LDS

    for (int kpos = 0; kpos < 9; ++kpos) {
      const int kh = kpos / 3;
      const int kw = kpos % 3;
      // ---- stage A chunk [128 o x 32 k] via Tensor Data Mover ----
      if (wid == 0) {
        tdm_load_2d_bf16(asBase, aggB + kpos * CIN + ic0,
                         /*tile_w=*/32, /*tile_h=*/COUT,
                         /*row_stride=*/KTOT);
        __builtin_amdgcn_s_wait_tensorcnt(0);
      }
      __syncthreads();                         // publish As + halo to all waves

      // ---- A fragment per ISA 16-bit A layout ----
      union { uint4 u[2]; v16bf v; } ua;
      {
        const int row = (wid << 4) + mlo;
        const int kA1 = hi * 8;
        const int kA2 = 16 + hi * 8;
        ua.u[0] = *reinterpret_cast<const uint4*>(&As[row][kA1]);
        ua.u[1] = *reinterpret_cast<const uint4*>(&As[row][kA2]);
      }
      // ---- 4 B fragments straight from halo + WMMA ----
      #pragma unroll
      for (int nsub = 0; nsub < 4; ++nsub) {
        const int n  = (nsub << 4) + mlo;      // column N (pixel)
        const int hy = (n >> 3) + kh;
        const int hx = (n & 7) + kw;
        const int kB = hi << 3;                // K 8h..8h+7 and 16+8h..16+8h+7 per half-wave
        union { uint4 u[2]; v16bf v; } ub;
        ub.u[0] = *reinterpret_cast<const uint4*>(&halo[hy][hx][kB]);
        ub.u[1] = *reinterpret_cast<const uint4*>(&halo[hy][hx][kB + 16]);
        acc[nsub] = __builtin_amdgcn_wmma_f32_16x16x32_bf16(
            false, ua.v, false, ub.v, (short)0, acc[nsub], false, false);
      }
      __syncthreads();                         // As reads done before next TDM
    }
  }

  // ---- epilogue: D layout -> out[b][o][y][x] ----
  #pragma unroll
  for (int nsub = 0; nsub < 4; ++nsub) {
    const int n = (nsub << 4) + mlo;
    const int y = th0 + (n >> 3);
    const int xc = tw0 + (n & 7);
    #pragma unroll
    for (int v = 0; v < 8; ++v) {
      const int o = (wid << 4) + v + (hi << 3);
      out[(((size_t)(b * COUT + o)) << 12) + ((size_t)y << 6) + xc] = acc[nsub][v];
    }
  }
}

// ---------------- launcher --------------------------------------------------
extern "C" void kernel_launch(void* const* d_in, const int* in_sizes, int n_in,
                              void* d_out, int out_size, void* d_ws, size_t ws_size,
                              hipStream_t stream) {
  const float* x      = (const float*)d_in[0];   // [32,128,64,64]
  const float* psa_w1 = (const float*)d_in[1];   // [33,128]
  const float* psa_w2 = (const float*)d_in[2];   // [512,33]
  const float* psa_b2 = (const float*)d_in[3];   // [512]
  const float* weight = (const float*)d_in[4];   // [4,128,128,3,3]
  const float* thresh = (const float*)d_in[5];   // [1,4,128,1,1,1]
  float* out = (float*)d_out;

  char* w = (char*)d_ws;
  __bf16* xb   = (__bf16*)w;                 w += (size_t)BS * HW * CIN * 2;    // 33.5 MB NHWC
  __bf16* agg  = (__bf16*)w;                 w += (size_t)BS * COUT * KTOT * 2; // 9.4 MB
  float* pooled = (float*)w;                 w += (size_t)BS * CIN * 4;
  float* attn   = (float*)w;                 w += (size_t)BS * PPAT * COUT * 4;
  float* mask   = (float*)w;                 w += 256;

  pool_kernel<<<BS * CIN, 256, 0, stream>>>(x, pooled);
  transpose_convert_kernel<<<BS * 64, 256, 0, stream>>>(x, xb);
  attention_kernel<<<BS, 128, 0, stream>>>(pooled, psa_w1, psa_w2, psa_b2, attn);
  mask_kernel<<<PPAT * 9, 256, 0, stream>>>(weight, thresh, mask);
  aggregate_kernel<<<BS * COUT, 256, 0, stream>>>(weight, attn, mask, agg);
  conv_wmma_kernel<<<BS * 64, 256, 0, stream>>>(xb, agg, out);
}
